// STARobertaSelfAttention_35330400976903
// MI455X (gfx1250) — matrix operand order repaired and checked by reading, not compiled, not run on hardware
//
#include <hip/hip_runtime.h>
#include <hip/hip_bf16.h>

// ---------------------------------------------------------------------------
// STARobertaSelfAttention for MI455X (gfx1250, wave32, WMMA).
// GEMMs run on v_wmma_f32_16x16x32_bf16 with 32x32 register tiling;
// the MLP-softmax runs on fp32 VALU with LDS reductions.
// ---------------------------------------------------------------------------

typedef __attribute__((ext_vector_type(16))) __bf16 v16bf;
typedef __attribute__((ext_vector_type(8)))  float  v8f;

#define Bsz 4
#define Ssz 512
#define Dsz 768
#define NH  12
#define DH  64
#define BH  (Bsz * NH)   // 48

// ---------------- helpers ----------------

__device__ __forceinline__ unsigned short f2bf(float f) {
  union { float f; unsigned u; } c; c.f = f;
  unsigned u = c.u;
  u += 0x7FFFu + ((u >> 16) & 1u);           // round-to-nearest-even
  return (unsigned short)(u >> 16);
}

// A fragment: 16x32 bf16 (MxK), row-major A[m][k], lda in elements.
// Lane l (0..15) = row m0+l; VGPR j holds K = 16*half + 2j, 2j+1.
__device__ __forceinline__ v16bf load_a(const unsigned short* A, int lda,
                                        int m0, int k0, int lane) {
  int half = (lane >> 4) & 1, l = lane & 15;
  const unsigned short* row = A + (size_t)(m0 + l) * lda + k0 + 16 * half;
  union { v16bf v; unsigned u[8]; } f;
#pragma unroll
  for (int j = 0; j < 4; ++j) {
    f.u[j]     = *(const unsigned*)(row + 2 * j);
    f.u[j + 4] = *(const unsigned*)(row + 8 + 2 * j);
  }
  return f.v;
}

// B fragment: 32x16 bf16 (KxN) supplied as Bt row-major Bt[n][k], ldb in elems.
// Lane l (0..15) = col n0+l; VGPR j holds K = 16*half + 2j, 2j+1.
__device__ __forceinline__ v16bf load_b(const unsigned short* Bt, int ldb,
                                        int n0, int k0, int lane) {
  int half = (lane >> 4) & 1, l = lane & 15;
  const unsigned short* row = Bt + (size_t)(n0 + l) * ldb + k0 + 16 * half;
  union { v16bf v; unsigned u[8]; } f;
#pragma unroll
  for (int j = 0; j < 8; ++j) f.u[j] = *(const unsigned*)(row + 2 * j);
  return f.v;
}

#define WMMA_BF16(A_, B_, C_)                                               \
  __builtin_amdgcn_wmma_f32_16x16x32_bf16(false, (A_), false, (B_),         \
                                          (short)0, (C_), false, false)

// ---------------- kernel 0: fp32 -> bf16 convert ----------------

__global__ void cvt_f32_bf16(const float* __restrict__ src,
                             unsigned short* __restrict__ dst, int n) {
  for (int i = blockIdx.x * blockDim.x + threadIdx.x; i < n;
       i += gridDim.x * blockDim.x)
    dst[i] = f2bf(src[i]);
}

// ---------------- kernel 1: fused QKV projection ----------------
// A = hsb [2048 x 768] bf16, Bt = W{q,k,v} [768 x 768] bf16.
// One wave per 32x16 output tile; (2 A + 3 B) loads feed 6 WMMAs.

__global__ void qkv_gemm(const unsigned short* __restrict__ hsb,
                         const unsigned short* __restrict__ wq,
                         const unsigned short* __restrict__ wk,
                         const unsigned short* __restrict__ wv,
                         const float* __restrict__ bq,
                         const float* __restrict__ bk,
                         const float* __restrict__ bv,
                         unsigned short* __restrict__ qb,
                         unsigned short* __restrict__ kb,
                         unsigned short* __restrict__ vtb) {
  int wave = blockIdx.x * (blockDim.x >> 5) + (threadIdx.x >> 5);
  int lane = threadIdx.x & 31;
  int mT = wave / 48, nT = wave % 48;       // 64 x 48 tiles of 32x16
  int m0 = mT * 32, n0 = nT * 16;

  v8f q0 = {}, q1 = {}, k0a = {}, k1a = {}, v0 = {}, v1 = {};
  for (int k0 = 0; k0 < Dsz; k0 += 32) {
    v16bf a0 = load_a(hsb, Dsz, m0,      k0, lane);
    v16bf a1 = load_a(hsb, Dsz, m0 + 16, k0, lane);
    v16bf b0 = load_b(wq, Dsz, n0, k0, lane);
    v16bf b1 = load_b(wk, Dsz, n0, k0, lane);
    v16bf b2 = load_b(wv, Dsz, n0, k0, lane);
    if (k0 + 32 < Dsz)   // global_prefetch_b8, WGP scope
      __builtin_prefetch((const void*)(hsb + (size_t)(m0 + (lane & 15)) * Dsz +
                                       k0 + 32), 0, 3);
    q0  = WMMA_BF16(a0, b0, q0);   q1  = WMMA_BF16(a1, b0, q1);
    k0a = WMMA_BF16(a0, b1, k0a);  k1a = WMMA_BF16(a1, b1, k1a);
    v0  = WMMA_BF16(a0, b2, v0);   v1  = WMMA_BF16(a1, b2, v1);
  }

  int half = (lane >> 4) & 1, l = lane & 15;
  int n = n0 + l, h = n >> 6, dh = n & 63;
  float bqv = bq[n], bkv = bk[n], bvv = bv[n];
#pragma unroll
  for (int mt = 0; mt < 2; ++mt) {
    const v8f& AQ = mt ? q1 : q0;
    const v8f& AK = mt ? k1a : k0a;
    const v8f& AV = mt ? v1 : v0;
#pragma unroll
    for (int r = 0; r < 8; ++r) {
      int m = m0 + 16 * mt + r + 8 * half;     // token index
      int b = m >> 9, s = m & 511;
      size_t qi = ((size_t)(b * NH + h) * Ssz + s) * DH + dh;
      qb[qi] = f2bf((AQ[r] + bqv) * 0.125f);   // fold 1/sqrt(64)
      kb[qi] = f2bf(AK[r] + bkv);
      vtb[((size_t)(b * NH + h) * DH + dh) * Ssz + s] = f2bf(AV[r] + bvv);
    }
  }
}

// ---------------- kernel 2: scores = Qs . K^T + mask ----------------
// One wave per 32x32 tile: (2 A + 2 B) loads feed 4 WMMAs.

__global__ void scores_gemm(const unsigned short* __restrict__ qb,
                            const unsigned short* __restrict__ kb,
                            const float* __restrict__ mask,
                            float* __restrict__ scores) {
  int wave = blockIdx.x * (blockDim.x >> 5) + (threadIdx.x >> 5);
  int lane = threadIdx.x & 31;
  int bh = wave >> 8;                        // 16*16 tiles of 32x32 per (b,h)
  int qT = (wave >> 4) & 15, kT = wave & 15;
  const unsigned short* A  = qb + (size_t)bh * Ssz * DH;
  const unsigned short* Bt = kb + (size_t)bh * Ssz * DH;
  int m0 = qT * 32, n0 = kT * 32;

  v8f c00 = {}, c01 = {}, c10 = {}, c11 = {};
#pragma unroll
  for (int k0 = 0; k0 < DH; k0 += 32) {
    v16bf a0 = load_a(A, DH, m0,      k0, lane);
    v16bf a1 = load_a(A, DH, m0 + 16, k0, lane);
    v16bf b0 = load_b(Bt, DH, n0,      k0, lane);
    v16bf b1 = load_b(Bt, DH, n0 + 16, k0, lane);
    c00 = WMMA_BF16(a0, b0, c00);  c01 = WMMA_BF16(a0, b1, c01);
    c10 = WMMA_BF16(a1, b0, c10);  c11 = WMMA_BF16(a1, b1, c11);
  }

  int half = (lane >> 4) & 1, l = lane & 15;
  int batch = bh / NH;
#pragma unroll
  for (int mt = 0; mt < 2; ++mt) {
#pragma unroll
    for (int nt = 0; nt < 2; ++nt) {
      const v8f& C = mt ? (nt ? c11 : c10) : (nt ? c01 : c00);
      int n = n0 + 16 * nt + l;
      float mv = mask[batch * Ssz + n];
#pragma unroll
      for (int r = 0; r < 8; ++r) {
        int m = m0 + 16 * mt + r + 8 * half;
        scores[((size_t)bh * Ssz + m) * Ssz + n] = C[r] + mv;
      }
    }
  }
}

// ---------------- kernel 3: fit_softmax (MLP softmax) ----------------
// One 256-thread block per row (b,h,q); each thread owns 2 columns.

__device__ __forceinline__ float mlp_inv(float x,
                                         const float* __restrict__ w1,
                                         const float* __restrict__ b1,
                                         const float* __restrict__ w2,
                                         float b2) {
  float acc = b2;
#pragma unroll
  for (int i = 0; i < 64; ++i) {
    float h = fmaxf(fmaf(x, w1[i], b1[i]), 0.0f);
    acc = fmaf(h, w2[i], acc);
  }
  return acc;
}

__global__ void fit_softmax_k(const float* __restrict__ scores,
                              unsigned short* __restrict__ probs,
                              const float* __restrict__ ew1,
                              const float* __restrict__ eb1,
                              const float* __restrict__ ew2,
                              const float* __restrict__ eb2,
                              const float* __restrict__ iw1,
                              const float* __restrict__ ib1,
                              const float* __restrict__ iw2,
                              const float* __restrict__ ib2) {
  __shared__ float red[256];
  int row = blockIdx.x;                      // bh*512 + q
  int bh = row >> 9, q = row & 511;
  int tid = threadIdx.x;
  const float* X = scores + (size_t)row * Ssz;
  // Faithful to torch: reference value for row q is scores[b,h,0,q].
  float tmax = scores[(size_t)bh * Ssz * Ssz + q];

  float tp0 = fminf(X[tid]       - (tmax - 1.0f), 3.0f);
  float tp1 = fminf(X[tid + 256] - (tmax - 1.0f), 3.0f);

  float a0 = eb2[0], a1 = eb2[0];
#pragma unroll
  for (int i = 0; i < 32; ++i) {
    float w = ew1[i], b = eb1[i], v = ew2[i];
    a0 = fmaf(fmaxf(fmaf(tp0, w, b), 0.0f), v, a0);
    a1 = fmaf(fmaxf(fmaf(tp1, w, b), 0.0f), v, a1);
  }
  float xu0 = (tp0 > -20.0f) ? a0 : 0.0f;
  float xu1 = (tp1 > -20.0f) ? a1 : 0.0f;

  // partition = sum(X_up)
  red[tid] = xu0 + xu1;
  __syncthreads();
  for (int s = 128; s > 0; s >>= 1) {
    if (tid < s) red[tid] += red[tid + s];
    __syncthreads();
  }
  float part = red[0];
  __syncthreads();

  float pinv = mlp_inv(part, iw1, ib1, iw2, ib2[0]);  // uniform -> scalarized
  xu0 *= pinv; xu1 *= pinv;

  // partition_p2 = sum(out)
  red[tid] = xu0 + xu1;
  __syncthreads();
  for (int s = 128; s > 0; s >>= 1) {
    if (tid < s) red[tid] += red[tid + s];
    __syncthreads();
  }
  float part2 = red[0];

  float pinv2 = mlp_inv(part2, iw1, ib1, iw2, ib2[0]);
  float scale = (part2 > 1.5f) ? pinv2 : 1.0f;

  probs[(size_t)row * Ssz + tid]       = f2bf(xu0 * scale);
  probs[(size_t)row * Ssz + tid + 256] = f2bf(xu1 * scale);
}

// ---------------- kernel 4: ctx = probs . V, fused output permute ----------
// One wave per 32x32 tile (DH=64 -> 2 N-tiles per head).

__global__ void ctx_gemm(const unsigned short* __restrict__ probs,
                         const unsigned short* __restrict__ vtb,
                         float* __restrict__ out) {
  int wave = blockIdx.x * (blockDim.x >> 5) + (threadIdx.x >> 5);
  int lane = threadIdx.x & 31;
  int bh = wave >> 5;                        // 16*2 tiles of 32x32 per (b,h)
  int mT = (wave >> 1) & 15, nT = wave & 1;
  const unsigned short* A  = probs + (size_t)bh * Ssz * Ssz;
  const unsigned short* Bt = vtb   + (size_t)bh * DH * Ssz;
  int m0 = mT * 32, n0 = nT * 32;

  v8f c00 = {}, c01 = {}, c10 = {}, c11 = {};
  for (int k0 = 0; k0 < Ssz; k0 += 32) {
    v16bf a0 = load_a(A, Ssz, m0,      k0, lane);
    v16bf a1 = load_a(A, Ssz, m0 + 16, k0, lane);
    v16bf b0 = load_b(Bt, Ssz, n0,      k0, lane);
    v16bf b1 = load_b(Bt, Ssz, n0 + 16, k0, lane);
    if (k0 + 32 < Ssz)
      __builtin_prefetch((const void*)(A + (size_t)(m0 + (lane & 15)) * Ssz +
                                       k0 + 32), 0, 3);
    c00 = WMMA_BF16(a0, b0, c00);  c01 = WMMA_BF16(a0, b1, c01);
    c10 = WMMA_BF16(a1, b0, c10);  c11 = WMMA_BF16(a1, b1, c11);
  }

  int half = (lane >> 4) & 1, l = lane & 15;
  int b = bh / NH, h = bh % NH;
#pragma unroll
  for (int mt = 0; mt < 2; ++mt) {
#pragma unroll
    for (int nt = 0; nt < 2; ++nt) {
      const v8f& C = mt ? (nt ? c11 : c10) : (nt ? c01 : c00);
      int n = n0 + 16 * nt + l;              // dh
#pragma unroll
      for (int r = 0; r < 8; ++r) {
        int m = m0 + 16 * mt + r + 8 * half; // q
        out[(((size_t)b * Ssz + m) * NH + h) * DH + n] = C[r];
      }
    }
  }
}

// ---------------- launch ----------------

extern "C" void kernel_launch(void* const* d_in, const int* in_sizes, int n_in,
                              void* d_out, int out_size, void* d_ws, size_t ws_size,
                              hipStream_t stream) {
  (void)in_sizes; (void)n_in; (void)out_size; (void)ws_size;
  const float* hs   = (const float*)d_in[0];
  const float* mask = (const float*)d_in[1];
  const float* Wq   = (const float*)d_in[2];
  const float* bq   = (const float*)d_in[3];
  const float* Wk   = (const float*)d_in[4];
  const float* bk   = (const float*)d_in[5];
  const float* Wv   = (const float*)d_in[6];
  const float* bv   = (const float*)d_in[7];
  const float* ew1  = (const float*)d_in[8];
  const float* eb1  = (const float*)d_in[9];
  const float* ew2  = (const float*)d_in[10];
  const float* eb2  = (const float*)d_in[11];
  const float* iw1  = (const float*)d_in[12];
  const float* ib1  = (const float*)d_in[13];
  const float* iw2  = (const float*)d_in[14];
  const float* ib2  = (const float*)d_in[15];

  char* ws = (char*)d_ws;
  constexpr size_t HSB  = 0;                                 // 2048*768 bf16
  constexpr size_t WQB  = HSB  + (size_t)2048 * 768 * 2;
  constexpr size_t WKB  = WQB  + (size_t)768 * 768 * 2;
  constexpr size_t WVB  = WKB  + (size_t)768 * 768 * 2;
  constexpr size_t QB   = WVB  + (size_t)768 * 768 * 2;      // [BH][S][DH] bf16
  constexpr size_t KB   = QB   + (size_t)BH * Ssz * DH * 2;
  constexpr size_t VTB  = KB   + (size_t)BH * Ssz * DH * 2;  // [BH][DH][S] bf16
  constexpr size_t SC   = VTB  + (size_t)BH * Ssz * DH * 2;  // [BH][S][S] f32
  constexpr size_t PRB  = SC   + (size_t)BH * Ssz * Ssz * 4; // [BH][S][S] bf16

  unsigned short* hsb = (unsigned short*)(ws + HSB);
  unsigned short* wqb = (unsigned short*)(ws + WQB);
  unsigned short* wkb = (unsigned short*)(ws + WKB);
  unsigned short* wvb = (unsigned short*)(ws + WVB);
  unsigned short* qb  = (unsigned short*)(ws + QB);
  unsigned short* kb  = (unsigned short*)(ws + KB);
  unsigned short* vtb = (unsigned short*)(ws + VTB);
  float*          sc  = (float*)(ws + SC);
  unsigned short* prb = (unsigned short*)(ws + PRB);

  cvt_f32_bf16<<<2048, 256, 0, stream>>>(hs, hsb, 2048 * 768);
  cvt_f32_bf16<<<1024, 256, 0, stream>>>(Wq, wqb, 768 * 768);
  cvt_f32_bf16<<<1024, 256, 0, stream>>>(Wk, wkb, 768 * 768);
  cvt_f32_bf16<<<1024, 256, 0, stream>>>(Wv, wvb, 768 * 768);

  // 64x48 tiles of 32x16, 1 wave/tile, 4 waves/block
  qkv_gemm<<<768, 128, 0, stream>>>(hsb, wqb, wkb, wvb, bq, bk, bv,
                                    qb, kb, vtb);
  // 48 * 16 * 16 tiles of 32x32
  scores_gemm<<<3072, 128, 0, stream>>>(qb, kb, mask, sc);
  // one block per (b,h,q) row
  fit_softmax_k<<<BH * Ssz, 256, 0, stream>>>(sc, prb, ew1, eb1, ew2, eb2,
                                              iw1, ib1, iw2, ib2);
  // 48 * 16 * 2 tiles of 32x32
  ctx_gemm<<<384, 128, 0, stream>>>(prb, vtb, (float*)d_out);
}
